// LearnedSimulator_24824910971472
// MI455X (gfx1250) — matrix operand order repaired and checked by reading, not compiled, not run on hardware
//
#include <hip/hip_runtime.h>

// ---------------------------------------------------------------------------
// Types
// ---------------------------------------------------------------------------
typedef __attribute__((ext_vector_type(16))) __bf16 v16bf;
typedef __attribute__((ext_vector_type(8)))  float  v8f;
typedef __attribute__((ext_vector_type(4)))  unsigned int u32x4;
typedef __attribute__((ext_vector_type(4)))  float  f32x4;

union ABfrag { v16bf v; u32x4 q[2]; };

#define DEVFN __device__ __forceinline__

DEVFN unsigned short f2bf(float x) {
    unsigned int u = __float_as_uint(x);
    u += 0x7FFFu + ((u >> 16) & 1u);       // round to nearest even
    return (unsigned short)(u >> 16);
}
DEVFN float bf2f(unsigned short b) {
    return __uint_as_float(((unsigned int)b) << 16);
}

constexpr int MODE_ENC  = 0;   // direct bf16 input (K1 cols), write Yout (LN)
constexpr int MODE_EDGE = 1;   // gather [e, v[s], v[r]]; e += msg; atomic agg
constexpr int MODE_NODE = 2;   // gather [v, agg]; v += upd
constexpr int MODE_DEC  = 3;   // direct from v (decoder staging)

// ---------------------------------------------------------------------------
// Stage input rows (128 rows x K1 bf16) into LDS
// ---------------------------------------------------------------------------
template<int K1, int MODE>
DEVFN void stage_in(unsigned short* Xs, int rowbase_g,
                    const unsigned short* Xin, const unsigned short* Vact,
                    const unsigned short* Eact, const float* Agg,
                    const int* senders, const int* receivers, int tid)
{
    constexpr int SEGS = K1 / 8;               // 16B chunks per row
    for (int idx = tid; idx < 128 * SEGS; idx += 256) {
        const int row = idx / SEGS;
        const int seg = idx - row * SEGS;
        const int rg  = rowbase_g + row;
        u32x4 d;
        if (MODE == MODE_ENC) {
            d = *(const u32x4*)(Xin + rg * K1 + seg * 8);
        } else if (MODE == MODE_EDGE) {
            const unsigned short* src;
            if (seg < 16)      src = Eact + rg * 128 + seg * 8;
            else if (seg < 32) src = Vact + senders[rg] * 128 + (seg - 16) * 8;
            else               src = Vact + receivers[rg] * 128 + (seg - 32) * 8;
            d = *(const u32x4*)src;
        } else if (MODE == MODE_NODE) {
            if (seg < 16) {
                d = *(const u32x4*)(Vact + rg * 128 + seg * 8);
            } else {
                const float* ap = Agg + rg * 128 + (seg - 16) * 8;
                f32x4 f0 = *(const f32x4*)ap;
                f32x4 f1 = *(const f32x4*)(ap + 4);
                union { u32x4 q; unsigned short s[8]; } t;
                t.s[0] = f2bf(f0.x); t.s[1] = f2bf(f0.y);
                t.s[2] = f2bf(f0.z); t.s[3] = f2bf(f0.w);
                t.s[4] = f2bf(f1.x); t.s[5] = f2bf(f1.y);
                t.s[6] = f2bf(f1.z); t.s[7] = f2bf(f1.w);
                d = t.q;
            }
        } else { // MODE_DEC
            d = *(const u32x4*)(Vact + rg * 128 + seg * 8);
        }
        *(u32x4*)(Xs + row * K1 + seg * 8) = d;
    }
}

// ---------------------------------------------------------------------------
// One GEMM layer: Xs (128 x K, bf16, LDS) @ Wsw (K x 128, pre-swizzled bf16)
// Wave computes a 16-row x (NCT*16)-col strip.
// A fragment lane-contiguous (k = 16h + s); B fragments lane-contiguous.
// ---------------------------------------------------------------------------
template<int K, int NCT>
DEVFN void gemm_tiles(const unsigned short* Xs, const unsigned short* Wsw,
                      int wave, int lane, v8f* acc)
{
    static_assert(K % 32 == 0, "K must be multiple of 32");
    const int m    = wave * 16 + (lane & 15);
    const int aoff = (lane < 16) ? 0 : 16;
    for (int kt = 0; kt < K / 32; ++kt) {
        ABfrag a;
        a.q[0] = *(const u32x4*)(Xs + m * K + kt * 32 + aoff);
        a.q[1] = *(const u32x4*)(Xs + m * K + kt * 32 + 8 + aoff);
#pragma unroll
        for (int ct = 0; ct < NCT; ++ct) {
            ABfrag b;
            const u32x4* bp =
                (const u32x4*)(Wsw + (((kt * 8 + ct) * 32 + lane) * 16));
            b.q[0] = bp[0];
            b.q[1] = bp[1];
            acc[ct] = __builtin_amdgcn_wmma_f32_16x16x32_bf16(
                false, a.v, false, b.v, (short)0, acc[ct], false, false);
        }
    }
}

// bias + relu + bf16 store to LDS (stride-128 next-layer input)
DEVFN void epilogue_relu(v8f* acc, const float* bias, unsigned short* Xnext,
                         int wave, int lane)
{
    const int colb = lane & 15;
    const int rsel = (lane < 16) ? 0 : 8;
#pragma unroll
    for (int ct = 0; ct < 8; ++ct) {
        const int col = ct * 16 + colb;
        const float bb = bias[col];
#pragma unroll
        for (int j = 0; j < 8; ++j) {
            float y = fmaxf(acc[ct][j] + bb, 0.0f);
            Xnext[(wave * 16 + j + rsel) * 128 + col] = f2bf(y);
        }
    }
}

// bias + LayerNorm (in-register, shfl reductions) + mode-specific output
template<int MODE>
DEVFN void epilogue_ln_out(v8f* acc, const float* bias,
                           const float* ln_g, const float* ln_b,
                           unsigned short* Eact, unsigned short* Vact,
                           float* Agg, unsigned short* Yout,
                           const int* receivers,
                           int rowbase_g, int wave, int lane)
{
    const int colb = lane & 15;
    const int rsel = (lane < 16) ? 0 : 8;
    float s[8], ss[8];
#pragma unroll
    for (int j = 0; j < 8; ++j) { s[j] = 0.0f; ss[j] = 0.0f; }
#pragma unroll
    for (int ct = 0; ct < 8; ++ct) {
        const float bb = bias[ct * 16 + colb];
#pragma unroll
        for (int j = 0; j < 8; ++j) {
            float y = acc[ct][j] + bb;
            acc[ct][j] = y;
            s[j]  += y;
            ss[j] += y * y;
        }
    }
    // reduce across the 16-lane group that holds each row
#pragma unroll
    for (int mask = 1; mask < 16; mask <<= 1) {
#pragma unroll
        for (int j = 0; j < 8; ++j) {
            s[j]  += __shfl_xor(s[j],  mask, 32);
            ss[j] += __shfl_xor(ss[j], mask, 32);
        }
    }
    float mu[8], rs[8];
#pragma unroll
    for (int j = 0; j < 8; ++j) {
        mu[j] = s[j] * (1.0f / 128.0f);
        float var = ss[j] * (1.0f / 128.0f) - mu[j] * mu[j];
        rs[j] = rsqrtf(var + 1e-5f);
    }
    int rowg[8], rcv[8];
#pragma unroll
    for (int j = 0; j < 8; ++j) {
        rowg[j] = rowbase_g + wave * 16 + j + rsel;
        if (MODE == MODE_EDGE) rcv[j] = receivers[rowg[j]];
    }
#pragma unroll
    for (int ct = 0; ct < 8; ++ct) {
        const int col = ct * 16 + colb;
        const float gg  = ln_g[col];
        const float bb2 = ln_b[col];
#pragma unroll
        for (int j = 0; j < 8; ++j) {
            const float y = (acc[ct][j] - mu[j]) * rs[j] * gg + bb2;
            const int idx = rowg[j] * 128 + col;
            if (MODE == MODE_ENC) {
                Yout[idx] = f2bf(y);
            } else if (MODE == MODE_EDGE) {
                Eact[idx] = f2bf(bf2f(Eact[idx]) + y);          // e += msg
                atomicAdd(Agg + rcv[j] * 128 + col, y);          // segment_sum
            } else if (MODE == MODE_NODE) {
                Vact[idx] = f2bf(bf2f(Vact[idx]) + y);           // v += upd
            }
        }
    }
}

// ---------------------------------------------------------------------------
// Fused 3-layer MLP (K1 -> 128 relu -> 128 relu -> 128) + LayerNorm + output
// ---------------------------------------------------------------------------
template<int K1, int MODE>
__global__ __launch_bounds__(256, 1)
void mlp3_kernel(const unsigned short* __restrict__ Xin,
                 unsigned short* __restrict__ Vact,
                 unsigned short* __restrict__ Eact,
                 float* __restrict__ Agg,
                 const int* __restrict__ senders,
                 const int* __restrict__ receivers,
                 const unsigned short* __restrict__ Wsw,
                 const float* __restrict__ b1,
                 const float* __restrict__ b2,
                 const float* __restrict__ b3,
                 const float* __restrict__ ln_g,
                 const float* __restrict__ ln_b,
                 unsigned short* __restrict__ Yout)
{
    extern __shared__ __align__(16) unsigned char smem[];
    unsigned short* X0 = (unsigned short*)smem;                 // K1-stride
    unsigned short* X1 = (unsigned short*)(smem + K1 * 128 * 2); // 128-stride

    const int tid  = threadIdx.x;
    const int wave = tid >> 5;
    const int lane = tid & 31;
    const int rowbase_g = blockIdx.x * 128;

    stage_in<K1, MODE>(X0, rowbase_g, Xin, Vact, Eact, Agg,
                       senders, receivers, tid);
    __syncthreads();

    v8f acc[8];
    const v8f vzero = {0.f, 0.f, 0.f, 0.f, 0.f, 0.f, 0.f, 0.f};

    // layer 1: X0 (K1) -> X1
#pragma unroll
    for (int i = 0; i < 8; ++i) acc[i] = vzero;
    gemm_tiles<K1, 8>(X0, Wsw, wave, lane, acc);
    epilogue_relu(acc, b1, X1, wave, lane);
    __syncthreads();   // everyone done reading X0 before layer-2 overwrites it

    // layer 2: X1 -> X0 (128-stride, wave-local rows)
#pragma unroll
    for (int i = 0; i < 8; ++i) acc[i] = vzero;
    gemm_tiles<128, 8>(X1, Wsw + K1 * 128, wave, lane, acc);
    epilogue_relu(acc, b2, X0, wave, lane);

    // layer 3: X0 (128-stride, wave-local rows) -> registers -> LN -> out
#pragma unroll
    for (int i = 0; i < 8; ++i) acc[i] = vzero;
    gemm_tiles<128, 8>(X0, Wsw + (K1 + 128) * 128, wave, lane, acc);
    epilogue_ln_out<MODE>(acc, b3, ln_g, ln_b, Eact, Vact, Agg, Yout,
                          receivers, rowbase_g, wave, lane);
}

// ---------------------------------------------------------------------------
// Decoder: v -> 128 relu -> 128 relu -> 2 ; then position integration
// ---------------------------------------------------------------------------
__global__ __launch_bounds__(256, 1)
void decoder_kernel(const unsigned short* __restrict__ Vact,
                    const unsigned short* __restrict__ Wsw,
                    const float* __restrict__ b1,
                    const float* __restrict__ b2,
                    const float* __restrict__ b3,
                    const float* __restrict__ acc_mean,
                    const float* __restrict__ acc_std,
                    const float* __restrict__ pos,
                    float* __restrict__ out)
{
    extern __shared__ __align__(16) unsigned char smem[];
    unsigned short* X0 = (unsigned short*)smem;
    unsigned short* X1 = (unsigned short*)(smem + 128 * 128 * 2);

    const int tid  = threadIdx.x;
    const int wave = tid >> 5;
    const int lane = tid & 31;
    const int rowbase_g = blockIdx.x * 128;

    stage_in<128, MODE_DEC>(X0, rowbase_g, nullptr, Vact, nullptr, nullptr,
                            nullptr, nullptr, tid);
    __syncthreads();

    v8f acc[8];
    const v8f vzero = {0.f, 0.f, 0.f, 0.f, 0.f, 0.f, 0.f, 0.f};

#pragma unroll
    for (int i = 0; i < 8; ++i) acc[i] = vzero;
    gemm_tiles<128, 8>(X0, Wsw, wave, lane, acc);
    epilogue_relu(acc, b1, X1, wave, lane);
    __syncthreads();

#pragma unroll
    for (int i = 0; i < 8; ++i) acc[i] = vzero;
    gemm_tiles<128, 8>(X1, Wsw + 128 * 128, wave, lane, acc);
    epilogue_relu(acc, b2, X0, wave, lane);

    // layer 3: single 16-wide column tile (cols 0..1 valid)
    v8f acc3[1];
    acc3[0] = vzero;
    gemm_tiles<128, 1>(X0, Wsw + 256 * 128, wave, lane, acc3);

    const int col  = lane & 15;
    const int rsel = (lane < 16) ? 0 : 8;
    if (col < 2) {
        const float am = acc_mean[col];
        const float as = acc_std[col];
#pragma unroll
        for (int j = 0; j < 8; ++j) {
            const int rg = rowbase_g + wave * 16 + j + rsel;
            const float acc_n = acc3[0][j] + b3[col];
            const float a = acc_n * as + am;
            const float p_last = pos[rg * 12 + 10 + col];  // [:, 5, :]
            const float p_prev = pos[rg * 12 +  8 + col];  // [:, 4, :]
            out[rg * 2 + col] = 2.0f * p_last - p_prev + a;
        }
    }
}

// ---------------------------------------------------------------------------
// Weight conversion + swizzle: f32 (Ksrc x ncols, row-major) -> bf16 fragment
// layout where each lane's 16 B-matrix elements are contiguous (2x b128 load).
// B layout: VGPR j holds K = koff + 2j, koff + 2j + 1 ; col = ct*16 + (l&15).
// ---------------------------------------------------------------------------
__global__ void swizzle_w_kernel(const float* __restrict__ W,
                                 unsigned short* __restrict__ out,
                                 int Kpad, int Ksrc, int ncols)
{
    const int tid = blockIdx.x * blockDim.x + threadIdx.x;
    const int total = (Kpad / 32) * 256;
    if (tid >= total) return;
    const int lane = tid & 31;
    const int ct   = (tid >> 5) & 7;
    const int kt   = tid >> 8;
    const int col  = ct * 16 + (lane & 15);
    const int koff = (lane < 16) ? 0 : 16;
    unsigned short* dst = out + tid * 16;
#pragma unroll
    for (int e = 0; e < 16; ++e) {
        const int k = kt * 32 + koff + e;
        const float v = (k < Ksrc && col < ncols) ? W[k * ncols + col] : 0.0f;
        dst[e] = f2bf(v);
    }
}

// ---------------------------------------------------------------------------
// Feature builders (pad to 32 bf16 columns for the K=32 encoder GEMM)
// ---------------------------------------------------------------------------
__global__ void node_features_kernel(const float* __restrict__ pos,
                                     const float* __restrict__ vmean,
                                     const float* __restrict__ vstd,
                                     unsigned short* __restrict__ NF, int n)
{
    const int i = blockIdx.x * blockDim.x + threadIdx.x;
    if (i >= n) return;
    const float* p = pos + i * 12;          // C=6, D=2
    unsigned short* o = NF + i * 32;
    int idx = 0;
#pragma unroll
    for (int t = 0; t < 5; ++t)
#pragma unroll
        for (int d = 0; d < 2; ++d) {
            const float vel = p[(t + 1) * 2 + d] - p[t * 2 + d];
            o[idx++] = f2bf((vel - vmean[d]) / vstd[d]);
        }
    const float cx = p[10], cy = p[11];
    o[10] = f2bf(fminf(cx / 0.05f, 1.0f));
    o[11] = f2bf(fminf(cy / 0.05f, 1.0f));
    o[12] = f2bf(fminf((1.0f - cx) / 0.05f, 1.0f));
    o[13] = f2bf(fminf((1.0f - cy) / 0.05f, 1.0f));
#pragma unroll
    for (idx = 14; idx < 32; ++idx) o[idx] = 0;
}

__global__ void edge_features_kernel(const float* __restrict__ pos,
                                     const int* __restrict__ senders,
                                     const int* __restrict__ receivers,
                                     unsigned short* __restrict__ EF, int e)
{
    const int i = blockIdx.x * blockDim.x + threadIdx.x;
    if (i >= e) return;
    const int s = senders[i], r = receivers[i];
    const float rx = (pos[s * 12 + 10] - pos[r * 12 + 10]) * 20.0f; // /0.05
    const float ry = (pos[s * 12 + 11] - pos[r * 12 + 11]) * 20.0f;
    const float dist = sqrtf(rx * rx + ry * ry);
    unsigned short* o = EF + i * 32;
    o[0] = f2bf(rx);
    o[1] = f2bf(ry);
    o[2] = f2bf(dist);
#pragma unroll
    for (int k = 3; k < 32; ++k) o[k] = 0;
}

// ---------------------------------------------------------------------------
// Host launcher
// ---------------------------------------------------------------------------
extern "C" void kernel_launch(void* const* d_in, const int* in_sizes, int n_in,
                              void* d_out, int out_size, void* d_ws, size_t ws_size,
                              hipStream_t stream)
{
    (void)in_sizes; (void)out_size; (void)ws_size;
    constexpr int N = 32768, E = 262144;

    auto F = [&](int i) -> const float* {
        return (const float*)d_in[(i >= 0 && i < n_in) ? i : 0];
    };
    const float* pos       = F(0);
    const int*   senders   = (const int*)d_in[(2 < n_in) ? 2 : 0];
    const int*   receivers = (const int*)d_in[(3 < n_in) ? 3 : 0];
    // JAX pytree leaf order (dict keys sorted):
    // 0..3 inputs, 4..9 dec.mlp, 10..17 edge_enc{ln,mlp}, 18..97 edge_mp[10],
    // 98..105 node_enc, 106..185 node_mp[10], 186..189 vel/acc stats
    const int EB0 = 18, NB0 = 106;

    // workspace carve-up
    char* w = (char*)d_ws;
    auto alloc = [&](size_t bytes) -> void* {
        void* r = (void*)w;
        w += (bytes + 255) & ~(size_t)255;
        return r;
    };
    unsigned short* v_bf = (unsigned short*)alloc((size_t)N * 128 * 2);
    unsigned short* e_bf = (unsigned short*)alloc((size_t)E * 128 * 2);
    float*          agg  = (float*)         alloc((size_t)N * 128 * 4);
    unsigned short* NF   = (unsigned short*)alloc((size_t)N * 32 * 2);
    unsigned short* EF   = (unsigned short*)alloc((size_t)E * 32 * 2);
    unsigned short* Wsw  = (unsigned short*)alloc((size_t)(384 + 256) * 128 * 2);

    const size_t SMEM_ENC  = 32  * 128 * 2 + 128 * 128 * 2;   //  40 KB
    const size_t SMEM_EDGE = 384 * 128 * 2 + 128 * 128 * 2;   // 128 KB
    const size_t SMEM_NODE = 256 * 128 * 2 + 128 * 128 * 2;   //  96 KB
    const size_t SMEM_DEC  = 128 * 128 * 2 + 128 * 128 * 2;   //  64 KB

    auto swz3 = [&](const float* W1, int K1src, int K1pad,
                    const float* W2, const float* W3, int n3cols) {
        swizzle_w_kernel<<<dim3(K1pad / 32), dim3(256), 0, stream>>>(
            W1, Wsw, K1pad, K1src, 128);
        swizzle_w_kernel<<<dim3(4), dim3(256), 0, stream>>>(
            W2, Wsw + K1pad * 128, 128, 128, 128);
        swizzle_w_kernel<<<dim3(4), dim3(256), 0, stream>>>(
            W3, Wsw + (K1pad + 128) * 128, 128, 128, n3cols);
    };

    // 1) features
    node_features_kernel<<<dim3(N / 256), dim3(256), 0, stream>>>(
        pos, F(186), F(187), NF, N);
    edge_features_kernel<<<dim3(E / 256), dim3(256), 0, stream>>>(
        pos, senders, receivers, EF, E);

    // 2) encoders (K padded 14->32 and 3->32)
    swz3(F(100), 14, 32, F(102), F(104), 128);                  // node_enc
    mlp3_kernel<32, MODE_ENC><<<dim3(N / 128), dim3(256), SMEM_ENC, stream>>>(
        NF, nullptr, nullptr, nullptr, nullptr, nullptr, Wsw,
        F(101), F(103), F(105), F(98), F(99), v_bf);
    swz3(F(12), 3, 32, F(14), F(16), 128);                      // edge_enc
    mlp3_kernel<32, MODE_ENC><<<dim3(E / 128), dim3(256), SMEM_ENC, stream>>>(
        EF, nullptr, nullptr, nullptr, nullptr, nullptr, Wsw,
        F(13), F(15), F(17), F(10), F(11), e_bf);

    // 3) message-passing rounds
    for (int m = 0; m < 10; ++m) {
        hipMemsetAsync(agg, 0, (size_t)N * 128 * 4, stream);
        const int eb = EB0 + 8 * m;   // ln_g, ln_b, W1,b1, W2,b2, W3,b3
        swz3(F(eb + 2), 384, 384, F(eb + 4), F(eb + 6), 128);
        mlp3_kernel<384, MODE_EDGE>
            <<<dim3(E / 128), dim3(256), SMEM_EDGE, stream>>>(
            nullptr, v_bf, e_bf, agg, senders, receivers, Wsw,
            F(eb + 3), F(eb + 5), F(eb + 7), F(eb), F(eb + 1), nullptr);
        const int nb = NB0 + 8 * m;
        swz3(F(nb + 2), 256, 256, F(nb + 4), F(nb + 6), 128);
        mlp3_kernel<256, MODE_NODE>
            <<<dim3(N / 128), dim3(256), SMEM_NODE, stream>>>(
            nullptr, v_bf, nullptr, agg, nullptr, nullptr, Wsw,
            F(nb + 3), F(nb + 5), F(nb + 7), F(nb), F(nb + 1), nullptr);
    }

    // 4) decoder + integration
    swz3(F(4), 128, 128, F(6), F(8), 2);
    decoder_kernel<<<dim3(N / 128), dim3(256), SMEM_DEC, stream>>>(
        v_bf, Wsw, F(5), F(7), F(9), F(188), F(189), pos, (float*)d_out);
}
